// RoRA_77309411983
// MI455X (gfx1250) — matrix operand order repaired and checked by reading, not compiled, not run on hardware
//
#include <hip/hip_runtime.h>
#include <hip/hip_bf16.h>

typedef __attribute__((ext_vector_type(16))) _Float16 v16h;
typedef __attribute__((ext_vector_type(8)))  float    v8f;

#define DIM   1024
#define RANK8 8
#define WAVES 2   // waves per block; each wave owns a 16-row tile

// ---------------------------------------------------------------------------
// Kernel 1: tiny prep.  B = [U V] (1024x8).
//   G = B^T B (8x8);  sigma = sigmoid(gate);  C^{-1} = (2/sigma) * (-J)
//   M = (C^{-1} - G)^{-1}  (Gauss-Jordan, partial pivot)
//   W = 2 B M  (1024x8), stored as f16 in workspace.
// ---------------------------------------------------------------------------
__global__ void rora_prep(const float* __restrict__ U,
                          const float* __restrict__ V,
                          const float* __restrict__ gate,
                          _Float16* __restrict__ Wh) {
  __shared__ float G[64];
  __shared__ float Minv[64];
  const int tid = threadIdx.x;           // 64 threads
  const int a = tid >> 3, b = tid & 7;

  float s = 0.f;
  for (int k = 0; k < DIM; ++k) {
    float Ba = (a < 4) ? U[k * 4 + a] : V[k * 4 + a - 4];
    float Bb = (b < 4) ? U[k * 4 + b] : V[k * 4 + b - 4];
    s += Ba * Bb;
  }
  G[tid] = s;
  __syncthreads();

  if (tid == 0) {
    float sig = 1.f / (1.f + expf(-gate[0]));
    float c = 2.f / sig;
    float Aug[8][16];
    for (int i = 0; i < 8; ++i)
      for (int j = 0; j < 8; ++j) {
        Aug[i][j]     = -G[i * 8 + j];
        Aug[i][j + 8] = (i == j) ? 1.f : 0.f;
      }
    for (int i = 0; i < 4; ++i) {        // + (2/sigma)*(-J)
      Aug[i][i + 4] -= c;
      Aug[i + 4][i] += c;
    }
    // Gauss-Jordan with partial pivoting
    for (int col = 0; col < 8; ++col) {
      int piv = col; float best = fabsf(Aug[col][col]);
      for (int r = col + 1; r < 8; ++r) {
        float v = fabsf(Aug[r][col]);
        if (v > best) { best = v; piv = r; }
      }
      if (piv != col)
        for (int j = 0; j < 16; ++j) {
          float t = Aug[col][j]; Aug[col][j] = Aug[piv][j]; Aug[piv][j] = t;
        }
      float inv = 1.f / Aug[col][col];
      for (int j = 0; j < 16; ++j) Aug[col][j] *= inv;
      for (int r = 0; r < 8; ++r) {
        if (r == col) continue;
        float f = Aug[r][col];
        for (int j = 0; j < 16; ++j) Aug[r][j] -= f * Aug[col][j];
      }
    }
    for (int i = 0; i < 8; ++i)
      for (int j = 0; j < 8; ++j) Minv[i * 8 + j] = Aug[i][j + 8];
  }
  __syncthreads();

  for (int k = tid; k < DIM; k += 64) {
    for (int aa = 0; aa < RANK8; ++aa) {
      float w = 0.f;
      for (int bb = 0; bb < RANK8; ++bb) {
        float Bb = (bb < 4) ? U[k * 4 + bb] : V[k * 4 + bb - 4];
        w += Bb * Minv[bb * 8 + aa];
      }
      Wh[k * 8 + aa] = (_Float16)(2.f * w);
    }
  }
}

// ---------------------------------------------------------------------------
// Kernel 2: y = x + (x B) W^T.  x tile is DMA'd to LDS once (async-to-LDS),
// then consumed twice: f16 A-operand for T = X*Bpad (32 WMMAs over K=1024),
// and exact f32 C-operand for Y = T*W^T + X per 16-col chunk.
// ---------------------------------------------------------------------------
__global__ __launch_bounds__(32 * WAVES) void
rora_main(const float* __restrict__ x,
          const float* __restrict__ U,
          const float* __restrict__ V,
          const _Float16* __restrict__ Wh,
          float* __restrict__ y) {
  // dynamic LDS: XL f32 [WAVES][16*1024] | BLh f16 [16][1024] | TL f32 [WAVES][128]
  extern __shared__ __align__(16) unsigned char smem[];
  float*    XL  = (float*)smem;
  _Float16* BLh = (_Float16*)(smem + (size_t)WAVES * 16 * DIM * 4);
  float*    TL  = (float*)(smem + (size_t)WAVES * 16 * DIM * 4 + (size_t)16 * DIM * 2);

  const int lane = threadIdx.x & 31;
  const int wave = threadIdx.x >> 5;
  const long tile = (long)blockIdx.x * WAVES + wave;
  const long r0 = tile * 16;
  const float* xt = x + r0 * DIM;
  float* XLw = XL + wave * 16 * DIM;

  // --- async DMA: contiguous 64KB x tile -> LDS (ASYNCcnt-tracked) ---
  {
    unsigned lds_base = (unsigned)(uintptr_t)XLw;
    for (int it = 0; it < (16 * DIM) / (32 * 4); ++it) {   // 128 x b128 per lane-set
      unsigned off = (unsigned)(it * 32 + lane) * 16u;
      unsigned long long gaddr = (unsigned long long)(uintptr_t)xt + off;
      asm volatile("global_load_async_to_lds_b128 %0, %1, off"
                   :: "v"(lds_base + off), "v"(gaddr)
                   : "memory");
    }
  }

  // --- cooperative stage of Bpad^T (f16) layout [n][k], cols >=8 zero ---
  for (int i = threadIdx.x; i < 16 * DIM; i += 32 * WAVES) {
    int n = i >> 10, k = i & (DIM - 1);
    float v = (n < 4) ? U[k * 4 + n] : ((n < 8) ? V[k * 4 + n - 4] : 0.f);
    BLh[i] = (_Float16)v;
  }
  asm volatile("s_wait_asynccnt 0x0" ::: "memory");
  __syncthreads();

  const int row = lane & 15;      // M row (phase1 A) / N col (B operands)
  const int g = lane >> 4;        // lane half-group
  const _Float16* Brow = &BLh[row * DIM];
  const float* Xrow = &XLw[row * DIM];

  // ---- Phase 1: accumulate T = X * Bpad over K = 1024 ----
  v8f c = {};
  for (int k0 = 0; k0 < DIM; k0 += 32) {
    v16h a, b;
    // A operand, contiguous K order matching B: VGPR i: K = g*16 + 2i
    #pragma unroll
    for (int i = 0; i < 8; ++i) {
      int kk = k0 + g * 16 + 2 * i;
      float2 p = *(const float2*)&Xrow[kk];
      a[2 * i]     = (_Float16)p.x;
      a[2 * i + 1] = (_Float16)p.y;
    }
    // B operand, contiguous K order matching A: lane col n=row, K = k0 + g*16 + 0..15 (contiguous 32B)
    b = *(const v16h*)&Brow[k0 + g * 16];
    c = __builtin_amdgcn_wmma_f32_16x16x32_f16(false, a, false, b,
                                               (short)0, c, false, false);
  }

  // spill T (16x8) through LDS to re-shape C-layout -> A-layout
  #pragma unroll
  for (int r = 0; r < 8; ++r)
    if ((lane & 15) < 8) TL[wave * 128 + (r + g * 8) * 8 + (lane & 15)] = c[r];
  __syncthreads();

  // A2: T as 16x32 A operand, only K=0..7 nonzero (g==0, VGPRs 0..3)
  v16h a2 = {};
  if (g == 0) {
    #pragma unroll
    for (int i = 0; i < 4; ++i) {
      a2[2 * i]     = (_Float16)TL[wave * 128 + row * 8 + 2 * i];
      a2[2 * i + 1] = (_Float16)TL[wave * 128 + row * 8 + 2 * i + 1];
    }
  }

  // ---- Phase 2: y_chunk = T * W^T + x_chunk, 16 cols at a time ----
  for (int n0 = 0; n0 < DIM; n0 += 16) {
    v16h b2 = {};
    if (g == 0) {                       // B2: K=0..7 valid -> elements 0..7
      const _Float16* wp = Wh + (n0 + row) * 8;
      #pragma unroll
      for (int i = 0; i < 8; ++i) b2[i] = wp[i];
    }
    v8f c2;                             // C: exact f32 x chunk from LDS, C layout
    #pragma unroll
    for (int r = 0; r < 8; ++r)
      c2[r] = XLw[(r + g * 8) * DIM + n0 + (lane & 15)];
    c2 = __builtin_amdgcn_wmma_f32_16x16x32_f16(false, a2, false, b2,
                                                (short)0, c2, false, false);
    #pragma unroll
    for (int r = 0; r < 8; ++r)
      y[(r0 + r + g * 8) * DIM + n0 + (lane & 15)] = c2[r];
  }
}

extern "C" void kernel_launch(void* const* d_in, const int* in_sizes, int n_in,
                              void* d_out, int out_size, void* d_ws, size_t ws_size,
                              hipStream_t stream) {
  const float* x    = (const float*)d_in[0];
  const float* U    = (const float*)d_in[1];
  const float* V    = (const float*)d_in[2];
  const float* gate = (const float*)d_in[3];
  float* y = (float*)d_out;
  _Float16* Wh = (_Float16*)d_ws;      // 1024*8 f16 = 16 KB scratch

  const long rows = (long)in_sizes[0] / DIM;      // 32768
  rora_prep<<<1, 64, 0, stream>>>(U, V, gate, Wh);

  const int rows_per_block = 16 * WAVES;          // 32
  const size_t smem = (size_t)WAVES * 16 * DIM * 4   // XL (f32 x tiles)
                    + (size_t)16 * DIM * 2           // BLh (f16 Bpad^T)
                    + (size_t)WAVES * 128 * 4;       // TL
  rora_main<<<(int)(rows / rows_per_block), 32 * WAVES, smem, stream>>>(x, U, V, Wh, y);
}
